// ActionPolicy_64716567216771
// MI455X (gfx1250) — matrix operand order repaired and checked by reading, not compiled, not run on hardware
//
#include <hip/hip_runtime.h>

// ===========================================================================
// ActionPolicy (Mamba diffusion head) for MI455X / gfx1250.
// Big GEMMs run on v_wmma_f32_16x16x32_f16 (f16 in, f32 accumulate).
// GEMM tile: 32(M) x 64(N) per wave -> 8 WMMAs / 32-K step, B frags reused 2x.
// Small-K ops (conv K=4, Wdt K=16, act-embed K=7) + SSM scan run on VALU.
// ===========================================================================

typedef __attribute__((ext_vector_type(16))) _Float16 v16h;
typedef __attribute__((ext_vector_type(8)))  float    v8f;

#define CDIV(a,b) (((a)+(b)-1)/(b))

union HFrag { _Float16 h[16]; v16h v; uint4 u[2]; };

__device__ __forceinline__ float siluf(float v) { return v / (1.0f + __expf(-v)); }

// ---------------------------------------------------------------------------
// f32 -> f16 weight conversion
// ---------------------------------------------------------------------------
__global__ void cvt_f16_k(const float* __restrict__ s, _Float16* __restrict__ d, int n) {
  int i = blockIdx.x * blockDim.x + threadIdx.x;
  if (i < n) d[i] = (_Float16)s[i];
}

// ---------------------------------------------------------------------------
// WMMA GEMM: C[M,N] = act( A[M,K] @ W[N,K]^T + bias[N] ) (+ R[M,N])
// A row r lives at: A + (r/aper)*aouter + (r%aper)*lda   (aper=1<<30 => plain lda)
// One wave per 32x64 output tile. Out-of-range columns are CLAMPED for loads
// (garbage lanes masked at store) so the K loop is branch-free / EXEC-stable.
// Requires: M%32==0, K%32==0, lda%4==0. N arbitrary.
// ---------------------------------------------------------------------------
__global__ __launch_bounds__(256) void gemm_wmma_k(
    const float* __restrict__ A, int lda, int aper, long aouter,
    const _Float16* __restrict__ W, const float* __restrict__ bias,
    float* C, int ldc, const float* R, int ldr,
    int M, int N, int K, int act)
{
  const int tn = (N + 63) >> 6;
  const int tm = M >> 5;
  const long tiles = (long)tm * tn;
  const long wid = (long)blockIdx.x * (blockDim.x >> 5) + (threadIdx.x >> 5);
  if (wid >= tiles) return;                 // wave-uniform: EXEC stays all-ones inside
  const int mt = (int)(wid % tm);           // m fast-varying: waves in a block share W tiles
  const int nt = (int)(wid / tm);
  const int m0 = mt << 5;
  const int n0 = nt << 6;
  const int lane = threadIdx.x & 31;
  const int rsel = lane & 15;
  const int ah = (lane >> 4) << 3;          // A K-half offset (0 / 8); also C row offset
  const int bh = (lane >> 4) << 3;          // B K-half offset (0 / 8)

  const int arow0 = m0 + rsel;
  const int arow1 = m0 + 16 + rsel;
  const float* ap0 = A + (long)(arow0 / aper) * aouter + (long)(arow0 % aper) * lda + ah;
  const float* ap1 = A + (long)(arow1 / aper) * aouter + (long)(arow1 % aper) * lda + ah;

  const int col0 = n0 + rsel;
  const bool v0 = (col0      < N);
  const bool v1 = (col0 + 16 < N);
  const bool v2 = (col0 + 32 < N);
  const bool v3 = (col0 + 48 < N);
  // clamped columns: loads always in-bounds, invalid lanes produce garbage
  // in their own (masked) output columns only.
  const int c0 = v0 ? col0      : N - 1;
  const int c1 = v1 ? col0 + 16 : N - 1;
  const int c2 = v2 ? col0 + 32 : N - 1;
  const int c3 = v3 ? col0 + 48 : N - 1;
  const _Float16* wq0 = W + (long)c0 * K + bh;
  const _Float16* wq1 = W + (long)c1 * K + bh;
  const _Float16* wq2 = W + (long)c2 * K + bh;
  const _Float16* wq3 = W + (long)c3 * K + bh;

  v8f a00, a01, a02, a03, a10, a11, a12, a13;
  {
    float b0 = bias ? bias[c0] : 0.0f;
    float b1 = bias ? bias[c1] : 0.0f;
    float b2 = bias ? bias[c2] : 0.0f;
    float b3 = bias ? bias[c3] : 0.0f;
    for (int i = 0; i < 8; ++i) {
      a00[i] = b0; a01[i] = b1; a02[i] = b2; a03[i] = b3;
      a10[i] = b0; a11[i] = b1; a12[i] = b2; a13[i] = b3;
    }
  }

  for (int k0 = 0; k0 < K; k0 += 32) {
    // ---- A fragments (f32 -> f16 convert on the fly) ----
    HFrag fa0, fa1;
#define ALOAD(frag, base)                                                     \
    {                                                                         \
      const float* p = (base) + k0;                                           \
      float4 f0 = *(const float4*)(p);                                        \
      float4 f1 = *(const float4*)(p + 4);                                    \
      float4 f2 = *(const float4*)(p + 16);                                   \
      float4 f3 = *(const float4*)(p + 20);                                   \
      frag.h[0]=(_Float16)f0.x;  frag.h[1]=(_Float16)f0.y;                    \
      frag.h[2]=(_Float16)f0.z;  frag.h[3]=(_Float16)f0.w;                    \
      frag.h[4]=(_Float16)f1.x;  frag.h[5]=(_Float16)f1.y;                    \
      frag.h[6]=(_Float16)f1.z;  frag.h[7]=(_Float16)f1.w;                    \
      frag.h[8]=(_Float16)f2.x;  frag.h[9]=(_Float16)f2.y;                    \
      frag.h[10]=(_Float16)f2.z; frag.h[11]=(_Float16)f2.w;                   \
      frag.h[12]=(_Float16)f3.x; frag.h[13]=(_Float16)f3.y;                   \
      frag.h[14]=(_Float16)f3.z; frag.h[15]=(_Float16)f3.w;                   \
    }
    ALOAD(fa0, ap0)
    ALOAD(fa1, ap1)
#undef ALOAD
    // ---- B fragments (pre-converted f16 weights, two 8-half runs per lane at k0+8h and k0+16+8h, same k order as A) ----
    HFrag fb0, fb1, fb2, fb3;
    fb0.u[0] = *(const uint4*)(wq0 + k0); fb0.u[1] = *(const uint4*)(wq0 + k0 + 16);
    fb1.u[0] = *(const uint4*)(wq1 + k0); fb1.u[1] = *(const uint4*)(wq1 + k0 + 16);
    fb2.u[0] = *(const uint4*)(wq2 + k0); fb2.u[1] = *(const uint4*)(wq2 + k0 + 16);
    fb3.u[0] = *(const uint4*)(wq3 + k0); fb3.u[1] = *(const uint4*)(wq3 + k0 + 16);

    a00 = __builtin_amdgcn_wmma_f32_16x16x32_f16(false, fa0.v, false, fb0.v, (short)0, a00, false, false);
    a01 = __builtin_amdgcn_wmma_f32_16x16x32_f16(false, fa0.v, false, fb1.v, (short)0, a01, false, false);
    a02 = __builtin_amdgcn_wmma_f32_16x16x32_f16(false, fa0.v, false, fb2.v, (short)0, a02, false, false);
    a03 = __builtin_amdgcn_wmma_f32_16x16x32_f16(false, fa0.v, false, fb3.v, (short)0, a03, false, false);
    a10 = __builtin_amdgcn_wmma_f32_16x16x32_f16(false, fa1.v, false, fb0.v, (short)0, a10, false, false);
    a11 = __builtin_amdgcn_wmma_f32_16x16x32_f16(false, fa1.v, false, fb1.v, (short)0, a11, false, false);
    a12 = __builtin_amdgcn_wmma_f32_16x16x32_f16(false, fa1.v, false, fb2.v, (short)0, a12, false, false);
    a13 = __builtin_amdgcn_wmma_f32_16x16x32_f16(false, fa1.v, false, fb3.v, (short)0, a13, false, false);
  }

  const int rowb0 = m0 + ah;                // lanes 0-15 -> rows m0..m0+7, lanes 16-31 -> +8
  const int rowb1 = m0 + 16 + ah;
#define STOREACC(accv, rowb, colv, validv)                                    \
  if (validv) {                                                               \
    for (int r = 0; r < 8; ++r) {                                             \
      float v = (accv)[r];                                                    \
      if (act == 1) v = siluf(v);                                             \
      long row = (rowb) + r;                                                  \
      if (R) v += R[row * (long)ldr + (colv)];                                \
      C[row * (long)ldc + (colv)] = v;                                        \
    }                                                                         \
  }
  STOREACC(a00, rowb0, col0,      v0)
  STOREACC(a01, rowb0, col0 + 16, v1)
  STOREACC(a02, rowb0, col0 + 32, v2)
  STOREACC(a03, rowb0, col0 + 48, v3)
  STOREACC(a10, rowb1, col0,      v0)
  STOREACC(a11, rowb1, col0 + 16, v1)
  STOREACC(a12, rowb1, col0 + 32, v2)
  STOREACC(a13, rowb1, col0 + 48, v3)
#undef STOREACC
}

// ---------------------------------------------------------------------------
// RMSNorm over dim=256, one wave per row (8 values per lane)
// ---------------------------------------------------------------------------
__global__ void rmsnorm_k(const float* __restrict__ x, const float* __restrict__ w,
                          float* __restrict__ o, int rows)
{
  int wid = (blockIdx.x * blockDim.x + threadIdx.x) >> 5;
  int lane = threadIdx.x & 31;
  if (wid >= rows) return;
  const float4* xr = (const float4*)(x + (long)wid * 256);
  float4 a = xr[lane * 2];
  float4 b = xr[lane * 2 + 1];
  float s = a.x*a.x + a.y*a.y + a.z*a.z + a.w*a.w
          + b.x*b.x + b.y*b.y + b.z*b.z + b.w*b.w;
  for (int off = 16; off > 0; off >>= 1) s += __shfl_xor(s, off, 32);
  float sc = rsqrtf(s * (1.0f / 256.0f) + 1e-5f);
  const float4* wv = (const float4*)w;
  float4 wa = wv[lane * 2], wb = wv[lane * 2 + 1];
  float4 oa = {a.x*sc*wa.x, a.y*sc*wa.y, a.z*sc*wa.z, a.w*sc*wa.w};
  float4 ob = {b.x*sc*wb.x, b.y*sc*wb.y, b.z*sc*wb.z, b.w*sc*wb.w};
  float4* orow = (float4*)(o + (long)wid * 256);
  orow[lane * 2]     = oa;
  orow[lane * 2 + 1] = ob;
}

// ---------------------------------------------------------------------------
// sinusoidal time-embedding base: t0[b, j] = sin/cos(sigma[b] * freq)
// ---------------------------------------------------------------------------
__global__ void timebase_k(const float* __restrict__ sigma, float* __restrict__ t0, int n) {
  int i = blockIdx.x * blockDim.x + threadIdx.x;
  if (i >= n) return;
  int b = i >> 8, j = i & 255;
  int jj = j & 127;
  float fr = __expf(-logf(10000.0f) * (float)jj * (1.0f / 128.0f));
  float ang = sigma[b] * fr;
  t0[i] = (j < 128) ? sinf(ang) : cosf(ang);
}

// action embed: x[b, 2+t, :] = noisy_actions[b,t,:] @ aew^T + aeb   (K=7)
__global__ void actemb_k(const float* __restrict__ na, const float* __restrict__ aew,
                         const float* __restrict__ aeb, float* __restrict__ x, int n) {
  int i = blockIdx.x * blockDim.x + threadIdx.x;
  if (i >= n) return;                              // n = 2048*10*256
  int c = i & 255; int t = (i >> 8) % 10; int b = i / 2560;
  const float* ap = na + ((long)b * 10 + t) * 7;
  const float* wp = aew + c * 7;
  float acc = aeb[c];
  #pragma unroll
  for (int k = 0; k < 7; ++k) acc += ap[k] * wp[k];
  x[((long)b * 12 + 2 + t) * 256 + c] = acc;
}

__global__ void posadd_k(float* __restrict__ x, const float* __restrict__ pos, long n) {
  long i = (long)blockIdx.x * blockDim.x + threadIdx.x;
  if (i >= n) return;
  int l = (int)((i >> 8) % 12);
  x[i] += pos[l * 256 + (i & 255)];
}

// ---------------------------------------------------------------------------
// causal depthwise conv (K=4) + SiLU : xs = xz[:, :512] -> xc
// ---------------------------------------------------------------------------
__global__ void conv_silu_k(const float* __restrict__ xz, const float* __restrict__ cw,
                            const float* __restrict__ cb, float* __restrict__ xc, long n) {
  long i = (long)blockIdx.x * blockDim.x + threadIdx.x;
  if (i >= n) return;                              // n = 24576*512
  int d = i & 511; long m = i >> 9; int t = (int)(m % 12); long base = m - t;
  float acc = cb[d];
  #pragma unroll
  for (int k = 0; k < 4; ++k) {
    int l = t - 3 + k;
    if (l >= 0) acc += xz[(base + l) * 1024 + d] * cw[d * 4 + k];
  }
  xc[i] = siluf(acc);
}

// dt projection (K=16) + softplus
__global__ void dtproj_k(const float* __restrict__ dbc, const float* __restrict__ Wdt,
                         const float* __restrict__ bdt, float* __restrict__ dt, long n) {
  long i = (long)blockIdx.x * blockDim.x + threadIdx.x;
  if (i >= n) return;                              // n = 24576*512
  long m = i >> 9; int d = (int)(i & 511);
  const float* din = dbc + m * 144;
  const float* w = Wdt + d * 16;
  float acc = bdt[d];
  #pragma unroll
  for (int r = 0; r < 16; ++r) acc += din[r] * w[r];
  dt[i] = (acc > 20.0f) ? acc : logf(1.0f + __expf(acc));
}

// ---------------------------------------------------------------------------
// selective-scan: per (b,d) keep h[64] in regs, B/C staged in LDS.
// writes y_final = (y + xc*D) * silu(z) into xz[:, :512] (xs half is dead).
// ---------------------------------------------------------------------------
__global__ __launch_bounds__(512) void ssm_scan_k(
    const float* __restrict__ dbc, const float* __restrict__ dt,
    const float* __restrict__ xc, float* __restrict__ xz,
    const float* __restrict__ A_log, const float* __restrict__ Dp)
{
  const int b = blockIdx.x;
  const int d = threadIdx.x;                       // 512 threads = D_INNER
  float as[64], h[64];
  #pragma unroll
  for (int s = 0; s < 64; ++s) { as[s] = -__expf(A_log[d * 64 + s]); h[s] = 0.0f; }
  __shared__ float sB[64], sC[64];
  for (int t = 0; t < 12; ++t) {
    long m = (long)b * 12 + t;
    if (d < 64)       sB[d]      = dbc[m * 144 + 16 + d];
    else if (d < 128) sC[d - 64] = dbc[m * 144 + 80 + (d - 64)];
    __syncthreads();
    float dtv = dt[m * 512 + d];
    float xv  = xc[m * 512 + d];
    float dx  = dtv * xv;
    float yv  = 0.0f;
    #pragma unroll
    for (int s = 0; s < 64; ++s) {
      float dA = __expf(dtv * as[s]);
      h[s] = h[s] * dA + dx * sB[s];
      yv += h[s] * sC[s];
    }
    float zv = xz[m * 1024 + 512 + d];
    xz[m * 1024 + d] = (yv + xv * Dp[d]) * siluf(zv);
    __syncthreads();
  }
}

// gated-MLP gate: g[m,j] = h[m,j] * silu(h[m,256+j])
__global__ void gate_k(const float* __restrict__ h, float* __restrict__ g, long n) {
  long i = (long)blockIdx.x * blockDim.x + threadIdx.x;
  if (i >= n) return;                              // n = 24576*256
  long r = i >> 8; int j = (int)(i & 255);
  float yv = h[r * 512 + j];
  float gv = h[r * 512 + 256 + j];
  g[i] = yv * siluf(gv);
}

// corr input: c = concat(z_t, error)
__global__ void concat_k(const float* __restrict__ zt, const float* __restrict__ err,
                         float* __restrict__ c, long n) {
  long i = (long)blockIdx.x * blockDim.x + threadIdx.x;
  if (i >= n) return;                              // n = 2048*512
  long b = i >> 9; int j = (int)(i & 511);
  c[i] = (j < 256) ? zt[b * 256 + j] : err[b * 256 + (j - 256)];
}

// out = velocity + corr_weight * delta  (identical flat layouts)
__global__ void combine_k(const float* __restrict__ vel, const float* __restrict__ delta,
                          const float* __restrict__ cw, float* __restrict__ out, int n) {
  int i = blockIdx.x * blockDim.x + threadIdx.x;
  if (i >= n) return;
  out[i] = vel[i] + cw[0] * delta[i];
}

// ===========================================================================
// Host orchestration
// ===========================================================================
enum {
  IN_ZT = 0, IN_ZNEXT, IN_ERR, IN_ACT, IN_SIGMA,
  P_AEB = 5, P_AEW, P_CB1, P_CB2, P_CB3, P_CWT,
  P_CW1, P_CW2, P_CW3, P_HB1, P_HB2, P_HW1, P_HW2,
  P_L0 = 18,   // per layer (15 leaves): A_log, D, Wdt, Win, Wout, Wx, bdt, conv_b, conv_w, W1, W2, b1, b2, norm1, norm2
  P_NORMF = 63, P_POS, P_SPB, P_SPW, P_TW1, P_TW2, P_TB1, P_TB2
};

extern "C" void kernel_launch(void* const* d_in, const int* in_sizes, int n_in,
                              void* d_out, int out_size, void* d_ws, size_t ws_size,
                              hipStream_t stream)
{
  (void)in_sizes; (void)n_in; (void)ws_size;
  const int ML = 24576;   // B * 12 tokens
  const int MH = 20480;   // B * 10 action tokens
  const long BIG = 1L << 30;

  float* ws   = (float*)d_ws;
  float* x    = ws;                    // [ML,256]
  float* xn   = ws + 6291456L;         // [ML,256]
  float* xz   = ws + 12582912L;        // [ML,1024]  (y overwrites cols 0..511; reused as hbuf)
  float* xc   = ws + 37748736L;        // [ML,512]   (reused as mlp-h, then vel)
  float* dbc  = ws + 50331648L;        // [ML,144]   (reused as delta)
  float* dtb  = ws + 53870592L;        // [ML,512]   (reused as gate buffer)
  float* tmp  = ws + 66453504L;        // [2048,512]
  float* tmp2 = ws + 67502080L;        // [2048,512]
  _Float16* wh = (_Float16*)(ws + 68550656L);

  // f16 weight mirrors
  _Float16* tW1b = wh + 0;
  _Float16* tW2b = wh + 65536;
  _Float16* spwb = wh + 131072;
  _Float16* hw1b = wh + 196608;
  _Float16* hw2b = wh + 262144;        // 7*256
  _Float16* cw1b = wh + 263936;
  _Float16* cw2b = wh + 526080;
  _Float16* cw3b = wh + 788224;        // 70*512
  _Float16* lwb  = wh + 824064;        // per layer: Win(262144) Wx(73728) Wout(131072) W1(131072) W2(65536)

  auto CVT = [&](int idx, _Float16* dst, int n) {
    cvt_f16_k<<<CDIV(n, 256), 256, 0, stream>>>((const float*)d_in[idx], dst, n);
  };
  auto GEMM = [&](const float* A, int lda, int aper, long aouter,
                  const _Float16* W, const float* bias, float* C, int ldc,
                  const float* R, int ldr, int M, int N, int K, int act) {
    int tn = (N + 63) / 64;
    long tiles = (long)(M / 32) * tn;
    gemm_wmma_k<<<(int)((tiles + 7) / 8), 256, 0, stream>>>(
        A, lda, aper, aouter, W, bias, C, ldc, R, ldr, M, N, K, act);
  };

  // ---- weight conversion -------------------------------------------------
  CVT(P_TW1, tW1b, 65536);  CVT(P_TW2, tW2b, 65536);
  CVT(P_SPW, spwb, 65536);  CVT(P_HW1, hw1b, 65536);  CVT(P_HW2, hw2b, 1792);
  CVT(P_CW1, cw1b, 262144); CVT(P_CW2, cw2b, 262144); CVT(P_CW3, cw3b, 35840);
  for (int l = 0; l < 3; ++l) {
    int base = P_L0 + l * 15;
    _Float16* lw = lwb + (long)l * 663552;
    CVT(base + 3, lw + 0,      262144);  // Win  [1024,256]
    CVT(base + 5, lw + 262144, 73728);   // Wx   [144,512]
    CVT(base + 4, lw + 335872, 131072);  // Wout [256,512]
    CVT(base + 9, lw + 466944, 131072);  // W1   [512,256]
    CVT(base + 10,lw + 598016, 65536);   // W2   [256,256]
  }

  // ---- embeddings --------------------------------------------------------
  timebase_k<<<CDIV(2048 * 256, 256), 256, 0, stream>>>((const float*)d_in[IN_SIGMA], tmp, 2048 * 256);
  GEMM(tmp, 256, (int)BIG, 0, tW1b, (const float*)d_in[P_TB1], tmp2, 256,
       nullptr, 0, 2048, 256, 256, 1);                                   // silu(temb @ W1^T + b1)
  GEMM(tmp2, 256, (int)BIG, 0, tW2b, (const float*)d_in[P_TB2], x + 0, 3072,
       nullptr, 0, 2048, 256, 256, 0);                                   // token 0
  GEMM((const float*)d_in[IN_ZNEXT], 256, (int)BIG, 0, spwb, (const float*)d_in[P_SPB],
       x + 256, 3072, nullptr, 0, 2048, 256, 256, 0);                    // token 1
  actemb_k<<<CDIV(2048 * 10 * 256, 256), 256, 0, stream>>>(
      (const float*)d_in[IN_ACT], (const float*)d_in[P_AEW], (const float*)d_in[P_AEB],
      x, 2048 * 10 * 256);                                               // tokens 2..11
  posadd_k<<<CDIV((long)ML * 256, 256), 256, 0, stream>>>(x, (const float*)d_in[P_POS], (long)ML * 256);

  // ---- layers ------------------------------------------------------------
  for (int l = 0; l < 3; ++l) {
    int base = P_L0 + l * 15;
    _Float16* Winb  = lwb + (long)l * 663552;
    _Float16* Wxb   = Winb + 262144;
    _Float16* Woutb = Winb + 335872;
    _Float16* W1b   = Winb + 466944;
    _Float16* W2b   = Winb + 598016;

    rmsnorm_k<<<CDIV(ML, 8), 256, 0, stream>>>(x, (const float*)d_in[base + 13], xn, ML);
    GEMM(xn, 256, (int)BIG, 0, Winb, nullptr, xz, 1024, nullptr, 0, ML, 1024, 256, 0);
    conv_silu_k<<<CDIV((long)ML * 512, 256), 256, 0, stream>>>(
        xz, (const float*)d_in[base + 8], (const float*)d_in[base + 7], xc, (long)ML * 512);
    GEMM(xc, 512, (int)BIG, 0, Wxb, nullptr, dbc, 144, nullptr, 0, ML, 144, 512, 0);
    dtproj_k<<<CDIV((long)ML * 512, 256), 256, 0, stream>>>(
        dbc, (const float*)d_in[base + 2], (const float*)d_in[base + 6], dtb, (long)ML * 512);
    ssm_scan_k<<<2048, 512, 0, stream>>>(
        dbc, dtb, xc, xz, (const float*)d_in[base + 0], (const float*)d_in[base + 1]);
    GEMM(xz, 1024, (int)BIG, 0, Woutb, nullptr, x, 256, x, 256, ML, 256, 512, 0);  // x += y@Wout^T

    rmsnorm_k<<<CDIV(ML, 8), 256, 0, stream>>>(x, (const float*)d_in[base + 14], xn, ML);
    GEMM(xn, 256, (int)BIG, 0, W1b, (const float*)d_in[base + 11], xc, 512,
         nullptr, 0, ML, 512, 256, 0);                                   // h = xn@W1^T + b1
    gate_k<<<CDIV((long)ML * 256, 256), 256, 0, stream>>>(xc, dtb, (long)ML * 256);
    GEMM(dtb, 256, (int)BIG, 0, W2b, (const float*)d_in[base + 12], x, 256,
         x, 256, ML, 256, 256, 0);                                       // x += g@W2^T + b2
  }

  // ---- head --------------------------------------------------------------
  rmsnorm_k<<<CDIV(ML, 8), 256, 0, stream>>>(x, (const float*)d_in[P_NORMF], xn, ML);
  float* hbuf = xz;   // reuse
  float* vel  = xc;   // reuse
  GEMM(xn + 512, 256, 10, 3072, hw1b, (const float*)d_in[P_HB1], hbuf, 256,
       nullptr, 0, MH, 256, 256, 1);                                     // silu(a@hw1^T + b1), rows = (b, t>=2)
  GEMM(hbuf, 256, (int)BIG, 0, hw2b, (const float*)d_in[P_HB2], vel, 7,
       nullptr, 0, MH, 7, 256, 0);                                       // velocity [B*10, 7]

  // ---- correction MLP ----------------------------------------------------
  concat_k<<<CDIV(2048 * 512, 256), 256, 0, stream>>>(
      (const float*)d_in[IN_ZT], (const float*)d_in[IN_ERR], tmp, 2048L * 512);
  GEMM(tmp, 512, (int)BIG, 0, cw1b, (const float*)d_in[P_CB1], tmp2, 512,
       nullptr, 0, 2048, 512, 512, 1);
  GEMM(tmp2, 512, (int)BIG, 0, cw2b, (const float*)d_in[P_CB2], tmp, 512,
       nullptr, 0, 2048, 512, 512, 1);
  float* delta = dbc; // reuse
  GEMM(tmp, 512, (int)BIG, 0, cw3b, (const float*)d_in[P_CB3], delta, 70,
       nullptr, 0, 2048, 70, 512, 0);

  combine_k<<<CDIV(out_size, 256), 256, 0, stream>>>(
      vel, delta, (const float*)d_in[P_CWT], (float*)d_out, out_size);
}
